// MambaBlock_40664750358979
// MI455X (gfx1250) — matrix operand order repaired and checked by reading, not compiled, not run on hardware
//
#include <hip/hip_runtime.h>
#include <math.h>

// ---------------------------------------------------------------------------
// Mamba block for MI455X (gfx1250, wave32, WMMA f32<-f16 16x16x32)
// f16 operands in GEMMs (one-time weight transpose+convert), f32 accum,
// double-buffered LDS pipeline: global loads of chunk c+1 overlap WMMAs of c.
// ---------------------------------------------------------------------------

typedef __attribute__((ext_vector_type(16))) _Float16 v16h;
typedef __attribute__((ext_vector_type(8)))  float    v8f;

#define DMODEL 1024
#define DINNER 2048
#define DSTATE 16
#define SEQL   1024
#define BATCH  2
#define MROWS  (BATCH * SEQL)   // 2048 flattened (b,l) rows

// LDS row stride in halfs: 32 data + 8 pad (spreads 16 lanes across banks)
constexpr int LDS_S = 40;

enum { EPI_BIAS = 0, EPI_SOFTPLUS = 1, EPI_RESID = 2 };

union FragU { uint4 u[2]; v16h v; };

// C[M][N] = A[M][K] @ Bt[N][K]^T + epilogue.
// A, Bt: f16 row-major (contiguous in K). 256 threads = 8 waves.
// Block tile 128x128, wave tile 64x32 (4x2 WMMA accums), K chunks of 32,
// 2-deep software pipeline (regs -> LDS ping-pong).
template<int EPI>
__global__ __launch_bounds__(256)
void gemm_f16_wmma(const _Float16* __restrict__ A, const _Float16* __restrict__ Bt,
                   const float* __restrict__ bias, const float* __restrict__ resid,
                   float* __restrict__ C, int M, int N, int K)
{
    __shared__ _Float16 As[2][128 * LDS_S];   // [m][k]
    __shared__ _Float16 Bs[2][128 * LDS_S];   // [n][k]

    const int tid  = threadIdx.x;
    const int wave = tid >> 5;
    const int lane = tid & 31;
    const int half = lane >> 4;
    const int l16  = lane & 15;
    const int wm   = wave >> 2;               // 0..1  (64 rows each)
    const int wn   = wave & 3;                // 0..3  (32 cols each)

    const int m0 = blockIdx.y * 128;
    const int n0 = blockIdx.x * 128;

    v8f acc[4][2] = {};

    // staging roles: 2 threads per row, 32 bytes (2 x uint4) each
    const int srow  = tid >> 1;               // 0..127
    const int spart = tid & 1;                // halfs [0..15] or [16..31]
    const int soff  = srow * LDS_S + spart * 16;
    const bool bvalid = (n0 + srow < N);      // loop-invariant (M % 128 == 0)

    // zero-fill out-of-range B rows once, in both buffers
    if (!bvalid) {
        const uint4 z = {0u, 0u, 0u, 0u};
        uint4* d0 = reinterpret_cast<uint4*>(&Bs[0][soff]);
        uint4* d1 = reinterpret_cast<uint4*>(&Bs[1][soff]);
        d0[0] = z; d0[1] = z; d1[0] = z; d1[1] = z;
    }

    const _Float16* aptr = A + (size_t)(m0 + srow) * K + spart * 16;
    const _Float16* bptr = Bt + (size_t)(n0 + srow) * K + spart * 16;

    // prologue: stage chunk 0 into buffer 0
    uint4 ga0 = reinterpret_cast<const uint4*>(aptr)[0];
    uint4 ga1 = reinterpret_cast<const uint4*>(aptr)[1];
    uint4 gb0 = {0u, 0u, 0u, 0u}, gb1 = {0u, 0u, 0u, 0u};
    if (bvalid) {
        gb0 = reinterpret_cast<const uint4*>(bptr)[0];
        gb1 = reinterpret_cast<const uint4*>(bptr)[1];
    }
    {
        uint4* da = reinterpret_cast<uint4*>(&As[0][soff]);
        da[0] = ga0; da[1] = ga1;
        if (bvalid) {
            uint4* db = reinterpret_cast<uint4*>(&Bs[0][soff]);
            db[0] = gb0; db[1] = gb1;
        }
    }
    __syncthreads();

    const int nch = K >> 5;
    for (int c = 0; c < nch; ++c) {
        const int cur = c & 1;
        const bool hasNext = (c + 1 < nch);

        // issue chunk c+1 global loads (latency hidden behind WMMAs below)
        if (hasNext) {
            const _Float16* an = aptr + (c + 1) * 32;
            ga0 = reinterpret_cast<const uint4*>(an)[0];
            ga1 = reinterpret_cast<const uint4*>(an)[1];
            if (bvalid) {
                const _Float16* bn = bptr + (c + 1) * 32;
                gb0 = reinterpret_cast<const uint4*>(bn)[0];
                gb1 = reinterpret_cast<const uint4*>(bn)[1];
            }
        }

        // A 16x32 f16 fragment: element j -> K = (j>=8?16:0) + half*8 + (j&7)
        FragU fa[4];
        #pragma unroll
        for (int ti = 0; ti < 4; ++ti) {
            const _Float16* ap = &As[cur][(wm * 64 + ti * 16 + l16) * LDS_S + half * 8];
            fa[ti].u[0] = *reinterpret_cast<const uint4*>(ap);
            fa[ti].u[1] = *reinterpret_cast<const uint4*>(ap + 16);
        }
        // B 32x16 f16 fragment: element j -> K = (j>=8?16:0) + half*8 + (j&7)
        FragU fb[2];
        #pragma unroll
        for (int tj = 0; tj < 2; ++tj) {
            const _Float16* bp = &Bs[cur][(wn * 32 + tj * 16 + l16) * LDS_S + half * 8];
            fb[tj].u[0] = reinterpret_cast<const uint4*>(bp)[0];
            fb[tj].u[1] = *reinterpret_cast<const uint4*>(bp + 16);
        }
        #pragma unroll
        for (int ti = 0; ti < 4; ++ti)
            #pragma unroll
            for (int tj = 0; tj < 2; ++tj)
                acc[ti][tj] = __builtin_amdgcn_wmma_f32_16x16x32_f16(
                    false, fa[ti].v, false, fb[tj].v,
                    (short)0, acc[ti][tj], false, false);

        // store staged chunk c+1 into the other buffer
        if (hasNext) {
            uint4* da = reinterpret_cast<uint4*>(&As[cur ^ 1][soff]);
            da[0] = ga0; da[1] = ga1;
            if (bvalid) {
                uint4* db = reinterpret_cast<uint4*>(&Bs[cur ^ 1][soff]);
                db[0] = gb0; db[1] = gb1;
            }
        }
        __syncthreads();
    }

    // epilogue: C/D layout: elem v -> row = 8*half + v, col = l16
    #pragma unroll
    for (int tj = 0; tj < 2; ++tj) {
        const int col = n0 + wn * 32 + tj * 16 + l16;
        if (col >= N) continue;
        const float bv = bias[col];
        #pragma unroll
        for (int ti = 0; ti < 4; ++ti) {
            #pragma unroll
            for (int v = 0; v < 8; ++v) {
                const int row = m0 + wm * 64 + ti * 16 + half * 8 + v;
                float val = acc[ti][tj][v] + bv;
                if (EPI == EPI_SOFTPLUS)
                    val = (val > 20.f) ? val : log1pf(__expf(val));
                if (EPI == EPI_RESID)
                    val += resid[(size_t)row * N + col];
                C[(size_t)row * N + col] = val;
            }
        }
    }
}

// One-time: W (K x N, f32) -> Wt (N x K, f16). 32x32 tiles via LDS.
__global__ __launch_bounds__(256)
void transpose_cvt_kernel(const float* __restrict__ W, _Float16* __restrict__ Wt,
                          int K, int N)
{
    __shared__ float t[32][33];
    const int k0 = blockIdx.x * 32, n0 = blockIdx.y * 32;
    const int c = threadIdx.x & 31, r = threadIdx.x >> 5;   // 8 rows / step
    #pragma unroll
    for (int i = 0; i < 4; ++i)
        t[r + i * 8][c] = W[(size_t)(k0 + r + i * 8) * N + n0 + c];
    __syncthreads();
    #pragma unroll
    for (int i = 0; i < 4; ++i)
        Wt[(size_t)(n0 + r + i * 8) * K + k0 + c] = (_Float16)t[c][r + i * 8];
}

// One-time: elementwise f32 -> f16 (n % 1024 == 0)
__global__ __launch_bounds__(256)
void cvt_f16_kernel(const float* __restrict__ src, _Float16* __restrict__ dst)
{
    const size_t i = ((size_t)blockIdx.x * 256 + threadIdx.x) * 4;
    const float4 v = *reinterpret_cast<const float4*>(src + i);
    union { _Float16 h[4]; uint2 u; } p;
    p.h[0] = (_Float16)v.x; p.h[1] = (_Float16)v.y;
    p.h[2] = (_Float16)v.z; p.h[3] = (_Float16)v.w;
    *reinterpret_cast<uint2*>(dst + i) = p.u;
}

// Depthwise causal conv (k=4) + SiLU; writes f16 x_conv.
// grid: (DINNER/256, SEQL/128, BATCH)
__global__ __launch_bounds__(256)
void conv_silu_kernel(const float* __restrict__ xz, const float* __restrict__ cw,
                      const float* __restrict__ cb, _Float16* __restrict__ xc)
{
    const int d  = blockIdx.x * 256 + threadIdx.x;
    const int b  = blockIdx.z;
    const int l0 = blockIdx.y * 128;
    const float w0 = cw[d * 4 + 0], w1 = cw[d * 4 + 1];
    const float w2 = cw[d * 4 + 2], w3 = cw[d * 4 + 3];
    const float bb = cb[d];
    const size_t base = (size_t)b * SEQL;

    auto X = [&](int l) -> float {
        return xz[(base + l) * (size_t)(2 * DINNER) + d];
    };
    float p0 = (l0 - 3 >= 0) ? X(l0 - 3) : 0.f;
    float p1 = (l0 - 2 >= 0) ? X(l0 - 2) : 0.f;
    float p2 = (l0 - 1 >= 0) ? X(l0 - 1) : 0.f;
    for (int l = l0; l < l0 + 128; ++l) {
        const float cur = X(l);
        float a = fmaf(w0, p0, fmaf(w1, p1, fmaf(w2, p2, fmaf(w3, cur, bb))));
        a = a / (1.f + __expf(-a));  // SiLU
        xc[(base + l) * (size_t)DINNER + d] = (_Float16)a;
        p0 = p1; p1 = p2; p2 = cur;
    }
}

// Selective scan: one lane per (b, d, s); 16-lane shfl_xor reduce over states.
// Fuses D_skip and SiLU(z) gating; writes f16 y. 256 blocks x 256 threads.
__global__ __launch_bounds__(256)
void scan_kernel(const float* __restrict__ xz, const _Float16* __restrict__ xcv,
                 const float* __restrict__ dlt, const float* __restrict__ xp,
                 const float* __restrict__ A_log, const float* __restrict__ Dsk,
                 _Float16* __restrict__ y)
{
    const int g = blockIdx.x * 256 + threadIdx.x;
    const int s = g & 15;
    const int d = (g >> 4) & (DINNER - 1);
    const int b = g >> 15;

    const float a   = -__expf(A_log[d * DSTATE + s]);
    const float dsk = Dsk[d];
    float h = 0.f;
    size_t row = (size_t)b * SEQL;
    for (int l = 0; l < SEQL; ++l, ++row) {
        const float de = dlt[row * DINNER + d];
        const float xv = (float)xcv[row * DINNER + d];
        const float Bv = xp[row * (2 * DSTATE) + s];
        const float Cv = xp[row * (2 * DSTATE) + DSTATE + s];
        const float dA = __expf(de * a);
        h = fmaf(dA, h, de * xv * Bv);
        float part = h * Cv;
        part += __shfl_xor(part, 1, 32);
        part += __shfl_xor(part, 2, 32);
        part += __shfl_xor(part, 4, 32);
        part += __shfl_xor(part, 8, 32);
        if (s == 0) {
            const float z  = xz[row * (size_t)(2 * DINNER) + DINNER + d];
            const float sz = z / (1.f + __expf(-z));
            y[row * DINNER + d] = (_Float16)((part + dsk * xv) * sz);
        }
    }
}

// torch-style LayerNorm: unbiased var, eps added to std. One row per block.
__global__ __launch_bounds__(256)
void layernorm_kernel(const float* __restrict__ r, const float* __restrict__ alpha,
                      const float* __restrict__ beta, float* __restrict__ out)
{
    __shared__ float red[256];
    const int tid = threadIdx.x;
    const size_t row = blockIdx.x;
    const float4 v = reinterpret_cast<const float4*>(r + row * DMODEL)[tid];

    float sum = v.x + v.y + v.z + v.w;
    red[tid] = sum; __syncthreads();
    for (int st = 128; st > 0; st >>= 1) {
        if (tid < st) red[tid] += red[tid + st];
        __syncthreads();
    }
    const float mean = red[0] * (1.f / DMODEL);
    __syncthreads();

    const float dx = v.x - mean, dy = v.y - mean, dz = v.z - mean, dw = v.w - mean;
    red[tid] = dx * dx + dy * dy + dz * dz + dw * dw; __syncthreads();
    for (int st = 128; st > 0; st >>= 1) {
        if (tid < st) red[tid] += red[tid + st];
        __syncthreads();
    }
    const float var = red[0] * (1.f / (DMODEL - 1));
    const float inv = 1.f / (sqrtf(var) + 1e-6f);

    const float4 al = reinterpret_cast<const float4*>(alpha)[tid];
    const float4 be = reinterpret_cast<const float4*>(beta)[tid];
    float4 o;
    o.x = al.x * dx * inv + be.x;
    o.y = al.y * dy * inv + be.y;
    o.z = al.z * dz * inv + be.z;
    o.w = al.w * dw * inv + be.w;
    reinterpret_cast<float4*>(out + row * DMODEL)[tid] = o;
}

extern "C" void kernel_launch(void* const* d_in, const int* in_sizes, int n_in,
                              void* d_out, int out_size, void* d_ws, size_t ws_size,
                              hipStream_t stream)
{
    (void)in_sizes; (void)n_in; (void)out_size; (void)ws_size;
    const float* x      = (const float*)d_in[0];
    const float* W_in   = (const float*)d_in[1];
    const float* b_in   = (const float*)d_in[2];
    const float* conv_w = (const float*)d_in[3];
    const float* conv_b = (const float*)d_in[4];
    const float* W_xp   = (const float*)d_in[5];
    const float* b_xp   = (const float*)d_in[6];
    const float* W_dt   = (const float*)d_in[7];
    const float* b_dt   = (const float*)d_in[8];
    const float* A_log  = (const float*)d_in[9];
    const float* D_skip = (const float*)d_in[10];
    const float* W_out  = (const float*)d_in[11];
    const float* b_out  = (const float*)d_in[12];
    const float* alpha  = (const float*)d_in[13];
    const float* beta   = (const float*)d_in[14];
    float* out = (float*)d_out;

    char* ws = (char*)d_ws;
    const size_t MiB = 1024 * 1024;
    float*     xz     = (float*)(ws);                 // 2048x4096 f32 (32 MiB)
    _Float16*  xconvh = (_Float16*)(ws + 32 * MiB);   // 2048x2048 f16 (8 MiB)
    float*     delta  = (float*)(ws + 40 * MiB);      // 2048x2048 f32 (16 MiB)
    float*     xp     = (float*)(ws + 56 * MiB);      // 2048x32   f32 (256 KiB)
    _Float16*  yh     = (_Float16*)(ws + 57 * MiB);   // 2048x2048 f16 (8 MiB)
    _Float16*  xh     = (_Float16*)(ws + 65 * MiB);   // 2048x1024 f16 (4 MiB)
    _Float16*  Wt_in  = (_Float16*)(ws + 69 * MiB);   // 4096x1024 f16 (8 MiB)
    _Float16*  Wt_xp  = (_Float16*)(ws + 77 * MiB);   // 32x2048   f16 (128 KiB)
    _Float16*  Wt_dt  = (_Float16*)(ws + 78 * MiB);   // 2048x2048 f16 (8 MiB)
    _Float16*  Wt_out = (_Float16*)(ws + 86 * MiB);   // 1024x2048 f16 (4 MiB)
    float*     r      = (float*)(ws);                 // 2048x1024, reuses xz

    // ---- one-time operand preparation (f16, weights transposed to [N][K]) ----
    transpose_cvt_kernel<<<dim3(DMODEL / 32, 2 * DINNER / 32), 256, 0, stream>>>(
        W_in, Wt_in, DMODEL, 2 * DINNER);
    transpose_cvt_kernel<<<dim3(DINNER / 32, 2 * DSTATE / 32), 256, 0, stream>>>(
        W_xp, Wt_xp, DINNER, 2 * DSTATE);
    transpose_cvt_kernel<<<dim3(DINNER / 32, DINNER / 32), 256, 0, stream>>>(
        W_dt, Wt_dt, DINNER, DINNER);
    transpose_cvt_kernel<<<dim3(DINNER / 32, DMODEL / 32), 256, 0, stream>>>(
        W_out, Wt_out, DINNER, DMODEL);
    cvt_f16_kernel<<<dim3(MROWS * DMODEL / 1024), 256, 0, stream>>>(x, xh);

    // 1) xz = x @ W_in + b_in          (M=2048, N=4096, K=1024)
    gemm_f16_wmma<EPI_BIAS><<<dim3(4096 / 128, MROWS / 128), 256, 0, stream>>>(
        xh, Wt_in, b_in, nullptr, xz, MROWS, 2 * DINNER, DMODEL);

    // 2) depthwise causal conv + SiLU -> x_conv (f16)
    conv_silu_kernel<<<dim3(DINNER / 256, SEQL / 128, BATCH), 256, 0, stream>>>(
        xz, conv_w, conv_b, xconvh);

    // 3) xp = x_conv @ W_xp + b_xp     (M=2048, N=32, K=2048)
    gemm_f16_wmma<EPI_BIAS><<<dim3(1, MROWS / 128), 256, 0, stream>>>(
        xconvh, Wt_xp, b_xp, nullptr, xp, MROWS, 2 * DSTATE, DINNER);

    // 4) delta = softplus(x_conv @ W_dt + b_dt)   (M=2048, N=2048, K=2048)
    gemm_f16_wmma<EPI_SOFTPLUS><<<dim3(DINNER / 128, MROWS / 128), 256, 0, stream>>>(
        xconvh, Wt_dt, b_dt, nullptr, delta, MROWS, DINNER, DINNER);

    // 5) selective scan (fused D_skip + SiLU(z) gate) -> y (f16)
    scan_kernel<<<dim3(256), 256, 0, stream>>>(
        xz, xconvh, delta, xp, A_log, D_skip, yh);

    // 6) r = y @ W_out + b_out + x     (M=2048, N=1024, K=2048)
    gemm_f16_wmma<EPI_RESID><<<dim3(DMODEL / 128, MROWS / 128), 256, 0, stream>>>(
        yh, Wt_out, b_out, x, r, MROWS, DMODEL, DINNER);

    // 7) LayerNorm -> out
    layernorm_kernel<<<dim3(MROWS), 256, 0, stream>>>(r, alpha, beta, out);
}
